// Net_25975962206615
// MI455X (gfx1250) — matrix operand order repaired and checked by reading, not compiled, not run on hardware
//
#include <hip/hip_runtime.h>
#include <hip/hip_bf16.h>

#define N_NODES  100000
#define N_EDGES  1600000
#define STATE_DIM 5
#define EDGE_FEAT 8
#define MAX_ITER 50
#define N_TILES  (N_EDGES / 16)          // 100000 exact

#define EDGE_BLOCKS    625
#define EDGE_WAVES     (EDGE_BLOCKS * 8) // 5000
#define TILES_PER_WAVE (N_TILES / EDGE_WAVES) // 20 exact

typedef __attribute__((ext_vector_type(16))) _Float16 v16h;
typedef __attribute__((ext_vector_type(8)))  float    v8f;

// ---- fast activations: single CDNA5 trans ops, branch-free fallbacks ----
__device__ __forceinline__ float fast_tanh(float x) {
#if __has_builtin(__builtin_amdgcn_tanhf)
    return __builtin_amdgcn_tanhf(x);        // v_tanh_f32
#else
    return 1.f - 2.f / (__expf(2.f * x) + 1.f);
#endif
}

__device__ __forceinline__ float fast_rcp(float x) {
#if __has_builtin(__builtin_amdgcn_rcpf)
    return __builtin_amdgcn_rcpf(x);         // v_rcp_f32
#else
    return 1.f / x;
#endif
}

// ---------------- init: copy initial state/old into ws, reset flags ----------------
__global__ void init_kernel(const float* __restrict__ si, const float* __restrict__ oi,
                            float* __restrict__ state, float* __restrict__ oldst,
                            int* __restrict__ flags) {
    int i = blockIdx.x * blockDim.x + threadIdx.x;
    if (i < N_NODES * STATE_DIM) { state[i] = si[i]; oldst[i] = oi[i]; }
    if (i == 0) { flags[0] = 0; flags[1] = 0; flags[2] = 0; } // cond, k, any
}

// ---------------- initial: zero new_state + convergence reduction ----------------
__global__ void prep_kernel(const float* __restrict__ state, const float* __restrict__ oldst,
                            float* __restrict__ newst, int* __restrict__ flags) {
    int i = blockIdx.x * blockDim.x + threadIdx.x;
    int pred = 0;
    if (i < N_NODES) {
        float s2 = 0.f;
        #pragma unroll
        for (int d = 0; d < STATE_DIM; ++d) {
            float df = state[i * STATE_DIM + d] - oldst[i * STATE_DIM + d];
            s2 += df * df;
            newst[i * STATE_DIM + d] = 0.f;
        }
        // dist = sqrt(s2 + 1e-11) > 0.01  <=>  s2 + 1e-11 > 1e-4
        pred = (s2 + 1e-11f > 1e-4f) ? 1 : 0;
    }
    if (__any(pred)) {
        if ((threadIdx.x & 31) == 0) atomicOr(&flags[2], 1);
    }
}

__global__ void cond_kernel(int* __restrict__ flags) {
    int any = flags[2];
    int k   = flags[1];
    int c   = (any != 0 && k < MAX_ITER) ? 1 : 0;
    flags[0] = c;          // cond for this step
    flags[1] = k + c;      // k += cond
    flags[2] = 0;          // reset reduction flag
}

// ------- fused: masked carry update + next-step reduction + zero new_state -------
__global__ void updprep_kernel(float* __restrict__ state, float* __restrict__ oldst,
                               float* __restrict__ newst, int* __restrict__ flags) {
    const int c = flags[0];
    int i = blockIdx.x * blockDim.x + threadIdx.x;
    int pred = 0;
    if (i < N_NODES) {
        float s2 = 0.f;
        #pragma unroll
        for (int d = 0; d < STATE_DIM; ++d) {
            const int idx = i * STATE_DIM + d;
            float sv = state[idx];
            float ov = oldst[idx];
            if (c) {
                ov = sv;
                sv = newst[idx];
                oldst[idx] = ov;
                state[idx] = sv;
            }
            float df = sv - ov;
            s2 += df * df;
            newst[idx] = 0.f;
        }
        pred = (s2 + 1e-11f > 1e-4f) ? 1 : 0;
    }
    if (__any(pred)) {
        if ((threadIdx.x & 31) == 0) atomicOr(&flags[2], 1);
    }
}

// ---------------- edge MLP: gather -> WMMA(13->15) -> tanh -> WMMA(15->5) -> tanh -> scatter ----------------
// Explicit 2-tile software pipeline: both WMMA1s back-to-back, both WMMA2s
// back-to-back; each tile's WMMA->VALU hazard window is filled by the sibling
// tile's tanh/cvt/DS work.
__global__ void __launch_bounds__(256) edge_kernel(
    const float* __restrict__ edge_feat, const int* __restrict__ edge_src,
    const int* __restrict__ arc_rows, const float* __restrict__ arc_vals,
    const float* __restrict__ W1, const float* __restrict__ b1,
    const float* __restrict__ W2, const float* __restrict__ b2,
    const float* __restrict__ state, float* __restrict__ newst,
    const int* __restrict__ flags)
{
    if (flags[0] == 0) return;                      // uniform: EXEC stays full for WMMA

    __shared__ _Float16 h1t[8][2][16 * 16];         // per-wave, one 16x16 f16 tile per pipeline slot

    const int lane = threadIdx.x & 31;
    const int wave = threadIdx.x >> 5;
    const int col  = lane & 15;                     // N (cols) for B/C/D; M (row) for A
    const int hi   = lane >> 4;

    // B1 fragment: W1 [13,15] row-major, padded to 32x16. elems 0..7 hold K=8*hi+elem.
    v16h bf1 = {};
    if (col < 15) {
        #pragma unroll
        for (int s = 0; s < 8; ++s) { const int k = 8 * hi + s; if (k < 13) bf1[s] = (_Float16)W1[k * 15 + col]; }
    }
    // B2 fragment: W2 [15,5] padded to 32x16.
    v16h bf2 = {};
    if (col < 5) {
        #pragma unroll
        for (int s = 0; s < 8; ++s) { const int k = 8 * hi + s; if (k < 15) bf2[s] = (_Float16)W2[k * 5 + col]; }
    }
    // Bias folded into accumulators: C layout -> lane holds column N=col in every VGPR.
    float bb1 = (col < 15) ? b1[col] : 0.f;
    float bb2 = (col < 5)  ? b2[col] : 0.f;
    v8f c1 = {}, c2 = {};
    #pragma unroll
    for (int r = 0; r < 8; ++r) { c1[r] = bb1; c2[r] = bb2; }

    const int gwave = blockIdx.x * 8 + wave;        // 0..4999, exactly 20 tiles each
    _Float16* t0 = &h1t[wave][0][0];
    _Float16* t1 = &h1t[wave][1][0];

    for (int i = 0; i < TILES_PER_WAVE; i += 2) {
        const int base0 = (gwave + (i    ) * EDGE_WAVES) * 16;
        const int base1 = (gwave + (i + 1) * EDGE_WAVES) * 16;

        // ---- stage 1: A fragments for both tiles ----
        // A layout: hi==0 holds K=0..7 (elems 0..7), K=16..23 (elems 8..15);
        //           hi==1 holds K=8..15 (elems 0..7), K=24..31 (elems 8..15).
        v16h a0 = {}, a1 = {};
        if (hi == 0) {
            const float* ef0 = edge_feat + (base0 + col) * EDGE_FEAT;
            const float* ef1 = edge_feat + (base1 + col) * EDGE_FEAT;
            #pragma unroll
            for (int k = 0; k < 8; ++k) { a0[k] = (_Float16)ef0[k]; a1[k] = (_Float16)ef1[k]; }
        } else {
            const float* sp0 = state + edge_src[base0 + col] * STATE_DIM;
            const float* sp1 = state + edge_src[base1 + col] * STATE_DIM;
            #pragma unroll
            for (int k = 0; k < STATE_DIM; ++k) { a0[k] = (_Float16)sp0[k]; a1[k] = (_Float16)sp1[k]; }
        }

        // ---- stage 2: layer-1 WMMAs (independent, back-to-back) ----
        v8f d1a = __builtin_amdgcn_wmma_f32_16x16x32_f16(
            false, a0, false, bf1, (short)0, c1, false, false);
        v8f d1b = __builtin_amdgcn_wmma_f32_16x16x32_f16(
            false, a1, false, bf1, (short)0, c1, false, false);

        // ---- stage 3: tanh + transposed store to LDS (both tiles) ----
        #pragma unroll
        for (int r = 0; r < 8; ++r) {
            const int m = r + 8 * hi;               // D: lane half selects row group
            t0[m * 16 + col] = (_Float16)fast_tanh(d1a[r]);
            t1[m * 16 + col] = (_Float16)fast_tanh(d1b[r]);
        }

        // ---- stage 4: A2 fragments (DS pipe is in-order within a wave) ----
        v16h a2a = {}, a2b = {};
        {
            const int m = col;                      // A row M = col
            const int kbase = 8 * hi;               // col 15 of h1 is exactly 0
            #pragma unroll
            for (int k = 0; k < 8; ++k) {
                a2a[k] = t0[m * 16 + kbase + k];
                a2b[k] = t1[m * 16 + kbase + k];
            }
        }

        // ---- stage 5: layer-2 WMMAs (independent, back-to-back) ----
        v8f d2a = __builtin_amdgcn_wmma_f32_16x16x32_f16(
            false, a2a, false, bf2, (short)0, c2, false, false);
        v8f d2b = __builtin_amdgcn_wmma_f32_16x16x32_f16(
            false, a2b, false, bf2, (short)0, c2, false, false);

        // ---- stage 6: tanh, scale by arc_vals, scatter-add into new_state ----
        if (col < STATE_DIM) {
            #pragma unroll
            for (int r = 0; r < 8; ++r) {
                const int m  = r + 8 * hi;
                const int e0 = base0 + m;
                const int e1 = base1 + m;
                float v0 = fast_tanh(d2a[r]) * arc_vals[e0];
                float v1 = fast_tanh(d2b[r]) * arc_vals[e1];
                atomicAdd(&newst[arc_rows[e0] * STATE_DIM + col], v0);
                atomicAdd(&newst[arc_rows[e1] * STATE_DIM + col], v1);
            }
        }
    }
}

// ---------------- readout head: tanh(5->10) -> softmax(10->7) ----------------
__global__ void out_kernel(const float* __restrict__ state,
                           const float* __restrict__ W3, const float* __restrict__ b3,
                           const float* __restrict__ W4, const float* __restrict__ b4,
                           float* __restrict__ out, const int* __restrict__ flags, int out_size) {
    int i = blockIdx.x * blockDim.x + threadIdx.x;
    if (i < N_NODES) {
        float s[STATE_DIM];
        #pragma unroll
        for (int d = 0; d < STATE_DIM; ++d) s[d] = state[i * STATE_DIM + d];
        float h[10];
        #pragma unroll
        for (int j = 0; j < 10; ++j) {
            float acc = b3[j];
            #pragma unroll
            for (int d = 0; d < STATE_DIM; ++d) acc += s[d] * W3[d * 10 + j];
            h[j] = fast_tanh(acc);
        }
        float l[7]; float mx = -1e30f;
        #pragma unroll
        for (int j = 0; j < 7; ++j) {
            float acc = b4[j];
            #pragma unroll
            for (int k = 0; k < 10; ++k) acc += h[k] * W4[k * 7 + j];
            l[j] = acc; mx = fmaxf(mx, acc);
        }
        float se = 0.f;
        #pragma unroll
        for (int j = 0; j < 7; ++j) { l[j] = __expf(l[j] - mx); se += l[j]; }
        float inv = fast_rcp(se);
        #pragma unroll
        for (int j = 0; j < 7; ++j) out[i * 7 + j] = l[j] * inv;
    }
    if (i == 0 && out_size > N_NODES * 7) out[N_NODES * 7] = (float)flags[1];
}

extern "C" void kernel_launch(void* const* d_in, const int* in_sizes, int n_in,
                              void* d_out, int out_size, void* d_ws, size_t ws_size,
                              hipStream_t stream) {
    const float* edge_feat = (const float*)d_in[0];
    const int*   edge_src  = (const int*)  d_in[1];
    const int*   arc_rows  = (const int*)  d_in[2];
    const float* arc_vals  = (const float*)d_in[3];
    const float* state_ini = (const float*)d_in[4];
    const float* old_ini   = (const float*)d_in[5];
    const float* W1 = (const float*)d_in[6];  const float* b1 = (const float*)d_in[7];
    const float* W2 = (const float*)d_in[8];  const float* b2 = (const float*)d_in[9];
    const float* W3 = (const float*)d_in[10]; const float* b3 = (const float*)d_in[11];
    const float* W4 = (const float*)d_in[12]; const float* b4 = (const float*)d_in[13];
    float* out = (float*)d_out;

    float* ws    = (float*)d_ws;
    float* state = ws;
    float* oldst = ws + (size_t)N_NODES * STATE_DIM;
    float* newst = ws + (size_t)2 * N_NODES * STATE_DIM;
    int*   flags = (int*)(ws + (size_t)3 * N_NODES * STATE_DIM);

    const int tN5 = N_NODES * STATE_DIM;
    init_kernel<<<(tN5 + 255) / 256, 256, 0, stream>>>(state_ini, old_ini, state, oldst, flags);
    prep_kernel<<<(N_NODES + 255) / 256, 256, 0, stream>>>(state, oldst, newst, flags);

    for (int it = 0; it < MAX_ITER; ++it) {
        cond_kernel<<<1, 1, 0, stream>>>(flags);
        edge_kernel<<<EDGE_BLOCKS, 256, 0, stream>>>(edge_feat, edge_src, arc_rows, arc_vals,
                                                     W1, b1, W2, b2, state, newst, flags);
        updprep_kernel<<<(N_NODES + 255) / 256, 256, 0, stream>>>(state, oldst, newst, flags);
    }

    out_kernel<<<(N_NODES + 255) / 256, 256, 0, stream>>>(state, W3, b3, W4, b4, out, flags, out_size);
}
